// MultiHeadSelfAttention_2740189135249
// MI455X (gfx1250) — matrix operand order repaired and checked by reading, not compiled, not run on hardware
//
#include <hip/hip_runtime.h>
#include <hip/hip_bf16.h>

typedef __attribute__((ext_vector_type(16))) __bf16 bf16x16;
typedef __attribute__((ext_vector_type(8)))  __bf16 bf16x8;
typedef __attribute__((ext_vector_type(4)))  __bf16 bf16x4;
typedef __attribute__((ext_vector_type(8)))  float  f32x8;

#define EMBED   768
#define HEADS   12
#define HDIM    64
#define BATCH   8
#define SEQ     1024
#define MTOT    (BATCH*SEQ)       /* 8192 */
#define QKV_OUT (3*EMBED)         /* 2304 */
#define KSTEPS  (EMBED/32)        /* 24 */
#define QSCALE  0.18033688011112042f   /* (1/sqrt(64)) * log2(e) */
#define LDSB_STRIDE 40            /* padded bf16 row stride: 80B -> conflict-free */

// ---- CDNA5 async global->LDS path (VGPR-bypassing, ASYNCcnt-tracked) --------
#if defined(__HIP_DEVICE_COMPILE__) && \
    __has_builtin(__builtin_amdgcn_global_load_async_to_lds_b128) && \
    __has_builtin(__builtin_amdgcn_s_wait_asynccnt)
#define HAVE_ASYNC_LDS 1
#else
#define HAVE_ASYNC_LDS 0
#endif

#if HAVE_ASYNC_LDS
// builtin expects pointers to 'int __attribute__((vector_size(16)))' in
// global (AS1) / LDS (AS3) address spaces (from hipcc diagnostic).
typedef int v4i_vs __attribute__((__vector_size__(16)));
typedef __attribute__((address_space(1))) v4i_vs as1_v4i;
typedef __attribute__((address_space(3))) v4i_vs as3_v4i;
#endif

static __device__ inline f32x8 zero8() {
    f32x8 z;
#pragma unroll
    for (int i = 0; i < 8; ++i) z[i] = 0.0f;
    return z;
}

static __device__ inline bf16x16 cat8(bf16x8 lo, bf16x8 hi) {
    bf16x16 r;
#pragma unroll
    for (int i = 0; i < 8; ++i) { r[i] = lo[i]; r[i + 8] = hi[i]; }
    return r;
}

// WMMA bf16: D = A(16x32) * B(32x16) + C(16x16 f32)
static __device__ inline f32x8 wmma_bf16(bf16x16 a, bf16x16 b, f32x8 c) {
    return __builtin_amdgcn_wmma_f32_16x16x32_bf16(false, a, false, b, (short)0, c,
                                                   false, false);
}

// A-fragment (16x32, rows m, contraction k) from row-major [*, ld] bf16 matrix.
// Lane holds row m = row0 + (lane&15); K pattern: 16 contiguous k at kb = lane<16?0:16.
static __device__ inline bf16x16 load_a_frag(const __bf16* base, int row0, int ld,
                                             int k0, int lane) {
    int r  = row0 + (lane & 15);
    int kb = (lane < 16) ? 0 : 16;
    const __bf16* p = base + (size_t)r * ld + k0 + kb;
    return cat8(*(const bf16x8*)p, *(const bf16x8*)(p + 8));
}

// B-fragment (32x16): B[k][n] = Mat[n][k] for row-major Mat[*, ld].
// Lane holds column n = n0 + (lane&15); 16 contiguous k at kb = lane<16?0:16.
static __device__ inline bf16x16 load_b_fragT(const __bf16* base, int n0, int ld,
                                              int k0, int lane) {
    int n  = n0 + (lane & 15);
    int kb = (lane < 16) ? 0 : 16;
    const __bf16* p = base + (size_t)n * ld + k0 + kb;
    return cat8(*(const bf16x8*)p, *(const bf16x8*)(p + 8));
}

// B-fragment from LDS tile stored [n_local][k0..31] with padded stride.
static __device__ inline bf16x16 load_b_lds(const __bf16* lds, int n0, int lane) {
    int row = n0 + (lane & 15);
    int kb  = (lane < 16) ? 0 : 16;
    const __bf16* p = lds + row * LDSB_STRIDE + kb;
    return cat8(*(const bf16x8*)p, *(const bf16x8*)(p + 8));
}

// Cooperative stage of a 128x32 bf16 weight tile into (padded) LDS.
// 256 threads x 32B each; async path issues global_load_async_to_lds_b128.
static __device__ inline void stage_tile(const __bf16* __restrict__ W, int nBase,
                                         int k0, __bf16* buf, int tid) {
    int row = tid >> 1, half = tid & 1;
    const __bf16* src = W + (size_t)(nBase + row) * EMBED + k0 + half * 16;
    __bf16* dst = buf + row * LDSB_STRIDE + half * 16;
#if HAVE_ASYNC_LDS
    __builtin_amdgcn_global_load_async_to_lds_b128((as1_v4i*)(src),
                                                   (as3_v4i*)(dst), 0, 0);
    __builtin_amdgcn_global_load_async_to_lds_b128((as1_v4i*)(src + 8),
                                                   (as3_v4i*)(dst + 8), 0, 0);
#else
    bf16x8 a = *(const bf16x8*)src, b = *(const bf16x8*)(src + 8);
    *(bf16x8*)dst = a; *(bf16x8*)(dst + 8) = b;
#endif
}

static __device__ inline void stage_wait() {
#if HAVE_ASYNC_LDS
    __builtin_amdgcn_s_wait_asynccnt(0);
#endif
    __syncthreads();
}

// ---------------------------------------------------------------- conversions
__global__ __launch_bounds__(256) void cvt_f32_bf16_v4(const float4* __restrict__ in,
                                                       bf16x4* __restrict__ out, int n4) {
    int i = blockIdx.x * 256 + threadIdx.x;
    if (i < n4) {
        float4 v = in[i];
        bf16x4 o;
        o[0] = (__bf16)v.x; o[1] = (__bf16)v.y; o[2] = (__bf16)v.z; o[3] = (__bf16)v.w;
        out[i] = o;
    }
}

// ---------------------------------------------------------------- QKV GEMM
// Y = X(8192x768) * Wqkv^T(768x2304) + b ; scatter to Q,K (B,H,N,D) and V^T (B,H,D,N).
__global__ __launch_bounds__(256) void qkv_gemm(const __bf16* __restrict__ X,
                                                const __bf16* __restrict__ W,
                                                const float*  __restrict__ bias,
                                                __bf16* __restrict__ Qo,
                                                __bf16* __restrict__ Ko,
                                                __bf16* __restrict__ Vt) {
    __shared__ __bf16 lds[2][128 * LDSB_STRIDE];   // double-buffered weight tile
    int wv = threadIdx.x >> 5, lane = threadIdx.x & 31;
    int nBase = blockIdx.x * 128;
    int mBase = blockIdx.y * 128 + wv * 16;

    f32x8 acc[8];
#pragma unroll
    for (int t = 0; t < 8; ++t) acc[t] = zero8();

    stage_tile(W, nBase, 0, lds[0], threadIdx.x);
    stage_wait();

    for (int kb = 0; kb < KSTEPS; ++kb) {
        int cur = kb & 1;
        if (kb + 1 < KSTEPS)
            stage_tile(W, nBase, (kb + 1) * 32, lds[1 - cur], threadIdx.x);
        bf16x16 afrag = load_a_frag(X, mBase, EMBED, kb * 32, lane);
        bf16x16 bfr[8];
#pragma unroll
        for (int t = 0; t < 8; ++t) bfr[t] = load_b_lds(lds[cur], t * 16, lane);
#pragma unroll
        for (int t = 0; t < 8; ++t) acc[t] = wmma_bf16(afrag, bfr[t], acc[t]);
        stage_wait();
    }

    int halfsel = (lane < 16) ? 0 : 8;
#pragma unroll
    for (int t = 0; t < 8; ++t) {
        int o = nBase + t * 16 + (lane & 15);
        float bv = bias[o];
        int which = o / EMBED, rem = o % EMBED;
        int h = rem >> 6, d = rem & 63;
#pragma unroll
        for (int r = 0; r < 8; ++r) {
            int m = mBase + r + halfsel;
            int b = m >> 10, tok = m & 1023;
            int bh = b * HEADS + h;
            float v = acc[t][r] + bv;
            if (which == 0)
                Qo[((size_t)bh * SEQ + tok) * HDIM + d] = (__bf16)(v * QSCALE);
            else if (which == 1)
                Ko[((size_t)bh * SEQ + tok) * HDIM + d] = (__bf16)v;
            else
                Vt[((size_t)bh * HDIM + d) * SEQ + tok] = (__bf16)v;
        }
    }
}

// ---------------------------------------------------------------- flash attention
// grid: (B*H, SEQ/128); each wave owns 16 query rows, streams 32-key tiles.
__global__ __launch_bounds__(256) void attn_kernel(const __bf16* __restrict__ Q,
                                                   const __bf16* __restrict__ K,
                                                   const __bf16* __restrict__ Vt,
                                                   __bf16* __restrict__ O) {
    __shared__ __bf16 plds[8 * 16 * LDSB_STRIDE];
    int wv = threadIdx.x >> 5, lane = threadIdx.x & 31;
    int bh = blockIdx.x;
    int q0 = blockIdx.y * 128 + wv * 16;
    const __bf16* Qb = Q  + (size_t)bh * SEQ * HDIM;
    const __bf16* Kb = K  + (size_t)bh * SEQ * HDIM;
    const __bf16* Vb = Vt + (size_t)bh * HDIM * SEQ;
    __bf16* pl = plds + wv * 16 * LDSB_STRIDE;
    int halfsel = (lane < 16) ? 0 : 8;

    // Q fragments for the whole row-strip (K-dim = 64 -> two k-steps), kept in VGPRs
    bf16x16 qf0 = load_a_frag(Qb, q0, HDIM, 0,  lane);
    bf16x16 qf1 = load_a_frag(Qb, q0, HDIM, 32, lane);

    f32x8 acc[4];
#pragma unroll
    for (int t = 0; t < 4; ++t) acc[t] = zero8();
    float mrow[8], lrow[8];
#pragma unroll
    for (int r = 0; r < 8; ++r) { mrow[r] = -__builtin_inff(); lrow[r] = 0.0f; }

    for (int kt = 0; kt < SEQ; kt += 32) {
        // S = Q * K^T for two 16-key column tiles (scores already in log2 domain)
        f32x8 s0 = zero8(), s1 = zero8();
        s0 = wmma_bf16(qf0, load_b_fragT(Kb, kt,      HDIM, 0,  lane), s0);
        s0 = wmma_bf16(qf1, load_b_fragT(Kb, kt,      HDIM, 32, lane), s0);
        s1 = wmma_bf16(qf0, load_b_fragT(Kb, kt + 16, HDIM, 0,  lane), s1);
        s1 = wmma_bf16(qf1, load_b_fragT(Kb, kt + 16, HDIM, 32, lane), s1);

        float alpha[8];
#pragma unroll
        for (int r = 0; r < 8; ++r) {
            float tmax = fmaxf(s0[r], s1[r]);
#pragma unroll
            for (int msk = 1; msk < 16; msk <<= 1)
                tmax = fmaxf(tmax, __shfl_xor(tmax, msk, 32));
            float nm = fmaxf(mrow[r], tmax);
            float al = __builtin_exp2f(mrow[r] - nm);
            float p0 = __builtin_exp2f(s0[r] - nm);
            float p1 = __builtin_exp2f(s1[r] - nm);
            float rs = p0 + p1;
#pragma unroll
            for (int msk = 1; msk < 16; msk <<= 1)
                rs += __shfl_xor(rs, msk, 32);
            lrow[r] = lrow[r] * al + rs;
            mrow[r] = nm;
            alpha[r] = al;
            // stage P (C-layout -> LDS) for A-layout reload
            int row = r + halfsel, col = lane & 15;
            pl[row * LDSB_STRIDE + col]      = (__bf16)p0;
            pl[row * LDSB_STRIDE + 16 + col] = (__bf16)p1;
        }

        // rescale accumulators by per-row alpha
#pragma unroll
        for (int t = 0; t < 4; ++t)
#pragma unroll
            for (int r = 0; r < 8; ++r) acc[t][r] *= alpha[r];

        // reload P as an A-fragment (16 rows x 32 keys)
        bf16x16 pfrag;
        {
            int row = lane & 15;
            int kb  = (lane < 16) ? 0 : 16;
            const __bf16* p = pl + row * LDSB_STRIDE + kb;
            pfrag = cat8(*(const bf16x8*)p, *(const bf16x8*)(p + 8));
        }
        // O += P * V  (4 d-tiles; V stored transposed so B-frags are contiguous)
#pragma unroll
        for (int t = 0; t < 4; ++t)
            acc[t] = wmma_bf16(pfrag, load_b_fragT(Vb, t * 16, SEQ, kt, lane), acc[t]);
    }

    // normalize and write (B, N, C) bf16 for the projection GEMM
    int b = bh / HEADS, h = bh % HEADS;
#pragma unroll
    for (int t = 0; t < 4; ++t) {
        int d = t * 16 + (lane & 15);
#pragma unroll
        for (int r = 0; r < 8; ++r) {
            int tok = q0 + r + halfsel;
            float v = acc[t][r] / lrow[r];
            O[((size_t)b * SEQ + tok) * EMBED + h * HDIM + d] = (__bf16)v;
        }
    }
}

// ---------------------------------------------------------------- output projection
__global__ __launch_bounds__(256) void proj_gemm(const __bf16* __restrict__ A,
                                                 const __bf16* __restrict__ W,
                                                 const float*  __restrict__ bias,
                                                 float* __restrict__ out) {
    __shared__ __bf16 lds[2][128 * LDSB_STRIDE];
    int wv = threadIdx.x >> 5, lane = threadIdx.x & 31;
    int nBase = blockIdx.x * 128;
    int mBase = blockIdx.y * 128 + wv * 16;

    f32x8 acc[8];
#pragma unroll
    for (int t = 0; t < 8; ++t) acc[t] = zero8();

    stage_tile(W, nBase, 0, lds[0], threadIdx.x);
    stage_wait();

    for (int kb = 0; kb < KSTEPS; ++kb) {
        int cur = kb & 1;
        if (kb + 1 < KSTEPS)
            stage_tile(W, nBase, (kb + 1) * 32, lds[1 - cur], threadIdx.x);
        bf16x16 afrag = load_a_frag(A, mBase, EMBED, kb * 32, lane);
        bf16x16 bfr[8];
#pragma unroll
        for (int t = 0; t < 8; ++t) bfr[t] = load_b_lds(lds[cur], t * 16, lane);
#pragma unroll
        for (int t = 0; t < 8; ++t) acc[t] = wmma_bf16(afrag, bfr[t], acc[t]);
        stage_wait();
    }

    int halfsel = (lane < 16) ? 0 : 8;
#pragma unroll
    for (int t = 0; t < 8; ++t) {
        int n = nBase + t * 16 + (lane & 15);
        float bv = bias[n];
#pragma unroll
        for (int r = 0; r < 8; ++r) {
            int m = mBase + r + halfsel;
            out[(size_t)m * EMBED + n] = acc[t][r] + bv;
        }
    }
}

// ---------------------------------------------------------------- launch
extern "C" void kernel_launch(void* const* d_in, const int* in_sizes, int n_in,
                              void* d_out, int out_size, void* d_ws, size_t ws_size,
                              hipStream_t stream) {
    const float* x     = (const float*)d_in[0];
    const float* Wqkv  = (const float*)d_in[1];
    const float* bqkv  = (const float*)d_in[2];
    const float* Wproj = (const float*)d_in[3];
    const float* bproj = (const float*)d_in[4];
    float* out = (float*)d_out;

    char* ws = (char*)d_ws;
    size_t off = 0;
    auto alloc = [&](size_t bytes) -> void* {
        void* p = ws + off;
        off += (bytes + 255) & ~(size_t)255;
        return p;
    };
    __bf16* xbf     = (__bf16*)alloc((size_t)MTOT * EMBED * 2);
    __bf16* wqkvbf  = (__bf16*)alloc((size_t)QKV_OUT * EMBED * 2);
    __bf16* wprojbf = (__bf16*)alloc((size_t)EMBED * EMBED * 2);
    __bf16* q       = (__bf16*)alloc((size_t)MTOT * EMBED * 2);
    __bf16* k       = (__bf16*)alloc((size_t)MTOT * EMBED * 2);
    __bf16* vt      = (__bf16*)alloc((size_t)MTOT * EMBED * 2);
    __bf16* attnbuf = (__bf16*)alloc((size_t)MTOT * EMBED * 2);
    (void)ws_size; (void)in_sizes; (void)n_in; (void)out_size;

    int n4;
    n4 = (MTOT * EMBED) / 4;
    cvt_f32_bf16_v4<<<(n4 + 255) / 256, 256, 0, stream>>>((const float4*)x, (bf16x4*)xbf, n4);
    n4 = (QKV_OUT * EMBED) / 4;
    cvt_f32_bf16_v4<<<(n4 + 255) / 256, 256, 0, stream>>>((const float4*)Wqkv, (bf16x4*)wqkvbf, n4);
    n4 = (EMBED * EMBED) / 4;
    cvt_f32_bf16_v4<<<(n4 + 255) / 256, 256, 0, stream>>>((const float4*)Wproj, (bf16x4*)wprojbf, n4);

    qkv_gemm<<<dim3(QKV_OUT / 128, MTOT / 128), 256, 0, stream>>>(xbf, wqkvbf, bqkv, q, k, vt);
    attn_kernel<<<dim3(BATCH * HEADS, SEQ / 128), 256, 0, stream>>>(q, k, vt, attnbuf);
    proj_gemm<<<dim3(EMBED / 128, MTOT / 128), 256, 0, stream>>>(attnbuf, wprojbf, bproj, out);
}
